// AtenMmQMixedSigni8_18940805775747
// MI455X (gfx1250) — 8-bit matrix operand order repaired and checked by reading, compiled, not run on hardware
//
#include <hip/hip_runtime.h>
#include <stdint.h>


typedef int v8i __attribute__((ext_vector_type(8)));

#define MDIM 4096
#define KDIM 4096
#define NDIM 4096
#define KT   (KDIM / 64)
#define MT   (MDIM / 16)
#define NT   (NDIM / 16)

#define CONST_TERM (-43253760)

#define STAGE_BYTES 24576

__global__ void zero_kernel(int* __restrict__ p, int n) {
    int i = blockIdx.x * blockDim.x + threadIdx.x;
    if (i < n) p[i] = 0;
}

__global__ __launch_bounds__(256) void pack_x_kernel(const int* __restrict__ x,
                                                     uint32_t* __restrict__ apk,
                                                     int* __restrict__ rowsum) {
    const int t      = threadIdx.x;
    const int lane   = t >> 3;
    const int v      = t & 7;
    const int tile_m = blockIdx.x >> 6;
    const int tile_k = blockIdx.x & (KT - 1);
    const int h      = lane >> 4;
    const int mloc   = lane & 15;
    const int m      = tile_m * 16 + mloc;
    const int k0     = tile_k * 64 + (v >> 1) * 16 + (v & 1) * 4 + h * 8;

    const int* __restrict__ src = x + (size_t)m * KDIM + k0;
    uint32_t w = 0;
    int      s = 0;
#pragma unroll
    for (int b = 0; b < 4; ++b) {
        int e = src[b];
        s += e;
        w |= (uint32_t)(e & 0xFF) << (8 * b);
    }
    apk[(size_t)blockIdx.x * 256 + t] = w;

    __shared__ int rs[16];
    if (t < 16) rs[t] = 0;
    __syncthreads();
    atomicAdd(&rs[mloc], s);
    __syncthreads();
    if (t < 16) atomicAdd(&rowsum[tile_m * 16 + t], rs[t]);
}

__global__ __launch_bounds__(256) void pack_y_kernel(const int* __restrict__ y,
                                                     uint32_t* __restrict__ bpk,
                                                     int* __restrict__ colsum) {
    const int t      = threadIdx.x;
    const int lane   = t >> 3;
    const int v      = t & 7;
    const int tile_k = blockIdx.x >> 8;
    const int tile_n = blockIdx.x & (NT - 1);
    const int g      = lane >> 4;
    const int nloc   = lane & 15;
    const int n      = tile_n * 16 + nloc;
    const int k0     = tile_k * 64 + (v >> 1) * 16 + (v & 1) * 4 + g * 8;

    const int* __restrict__ src = y + (size_t)k0 * NDIM + n;
    uint32_t w = 0;
    int      s = 0;
#pragma unroll
    for (int b = 0; b < 4; ++b) {
        int e = src[(size_t)b * NDIM];
        s += e;
        w |= (uint32_t)(e & 0xFF) << (8 * b);
    }
    bpk[(size_t)blockIdx.x * 256 + t] = w;

    __shared__ int cs[16];
    if (t < 16) cs[t] = 0;
    __syncthreads();
    atomicAdd(&cs[nloc], s);
    __syncthreads();
    if (t < 16) atomicAdd(&colsum[tile_n * 16 + t], cs[t]);
}

__device__ __forceinline__ void async_copy16(uint32_t lds_addr, uint64_t gaddr) {
    asm volatile("global_load_async_to_lds_b128 %0, %1, off"
                 :: "v"(lds_addr), "v"(gaddr) : "memory");
}
__device__ __forceinline__ void wait_async0() {
    asm volatile("s_wait_asynccnt 0" ::: "memory");
}

__global__ __launch_bounds__(256) void gemm_iu8_kernel(const uint32_t* __restrict__ apk,
                                                       const uint32_t* __restrict__ bpk,
                                                       const int* __restrict__ rowsum,
                                                       const int* __restrict__ colsum,
                                                       float* __restrict__ out) {
    __shared__ __align__(128) uint8_t smem[2 * STAGE_BYTES];

    const int t    = threadIdx.x;
    const int wave = t >> 5;
    const int lane = t & 31;
    const int wm   = wave & 3;
    const int wn   = wave >> 2;
    const int bm   = blockIdx.x;
    const int bn   = blockIdx.y;

    const int sub    = t >> 6;
    const int within = (t & 63) * 16;

    const uint32_t lds0  = (uint32_t)(size_t)&smem[0];
    const uint64_t apk_b = (uint64_t)(size_t)apk;
    const uint64_t bpk_b = (uint64_t)(size_t)bpk;

    auto issue_stage = [&](int tk, uint32_t ldsbase) {
#pragma unroll
        for (int i = 0; i < 4; ++i) {
            const int tile = sub + i * 4;
            async_copy16(ldsbase + tile * 1024 + within,
                         apk_b + ((uint64_t)((bm * 16 + tile) * KT + tk) << 10) + within);
        }
#pragma unroll
        for (int i = 0; i < 2; ++i) {
            const int tile = sub + i * 4;
            async_copy16(ldsbase + 16384 + tile * 1024 + within,
                         bpk_b + ((uint64_t)(tk * NT + bn * 8 + tile) << 10) + within);
        }
    };

    v8i acc[4][4] = {};

    issue_stage(0, lds0);

    for (int s = 0; s < KT; ++s) {
        const int      p   = s & 1;
        const uint8_t* buf = smem + p * STAGE_BYTES;

        wait_async0();
        __syncthreads();

        if (s + 1 < KT) issue_stage(s + 1, lds0 + (1 - p) * STAGE_BYTES);

        const uint8_t* aL = buf + (wm * 4) * 1024 + lane * 32;
        const uint8_t* bL = buf + 16384 + (wn * 4) * 1024 + lane * 32;
        v8i a[4], b[4];
#pragma unroll
        for (int i = 0; i < 4; ++i) a[i] = *(const v8i*)(aL + i * 1024);
#pragma unroll
        for (int j = 0; j < 4; ++j) b[j] = *(const v8i*)(bL + j * 1024);

#pragma unroll
        for (int i = 0; i < 4; ++i)
#pragma unroll
            for (int j = 0; j < 4; ++j)
                acc[i][j] = __builtin_amdgcn_wmma_i32_16x16x64_iu8(
                    true, a[i], false, b[j], acc[i][j], false, false);

        __syncthreads();
    }

    const int h  = lane >> 4;
    const int nl = lane & 15;
#pragma unroll
    for (int i = 0; i < 4; ++i) {
        const int mbase = (bm * 16 + wm * 4 + i) * 16 + 8 * h;
        int rs[8];
#pragma unroll
        for (int r = 0; r < 8; ++r) rs[r] = rowsum[mbase + r];
#pragma unroll
        for (int j = 0; j < 4; ++j) {
            const int n  = (bn * 8 + wn * 4 + j) * 16 + nl;
            const int cs = colsum[n];
#pragma unroll
            for (int r = 0; r < 8; ++r) {
                const int tot = acc[i][j][r] + 66 * cs - 160 * rs[r] + CONST_TERM;
                out[(size_t)(mbase + r) * NDIM + n] = 7.5e-4f * (float)tot;
            }
        }
    }
}

extern "C" void kernel_launch(void* const* d_in, const int* in_sizes, int n_in,
                              void* d_out, int out_size, void* d_ws, size_t ws_size,
                              hipStream_t stream) {
    const int* x   = (const int*)d_in[0];
    const int* y   = (const int*)d_in[1];
    float*     out = (float*)d_out;

    char*     ws     = (char*)d_ws;
    uint32_t* apk    = (uint32_t*)ws;
    uint32_t* bpk    = (uint32_t*)(ws + ((size_t)16 << 20));
    int*      rowsum = (int*)(ws + ((size_t)32 << 20));
    int*      colsum = rowsum + MDIM;

    zero_kernel<<<(MDIM + NDIM + 255) / 256, 256, 0, stream>>>(rowsum, MDIM + NDIM);

    pack_x_kernel<<<MT * KT, 256, 0, stream>>>(x, apk, rowsum);
    pack_y_kernel<<<KT * NT, 256, 0, stream>>>(y, bpk, colsum);

    dim3 grid(MDIM / 256, NDIM / 128);
    gemm_iu8_kernel<<<grid, 256, 0, stream>>>(apk, bpk, rowsum, colsum, out);
}
